// GAU_63934883168317
// MI455X (gfx1250) — compile-verified, matrix operand order checked by reading, not run on hardware
//
#include <hip/hip_runtime.h>
#include <hip/hip_bf16.h>
#include <math.h>

// ---------------------------------------------------------------------------
// GAU block on MI455X (gfx1250, wave32). All GEMMs via v_wmma_f32_16x16x32_f16.
// Wave tile: 32(M) x 64(N) -> 8 accumulators, 8 WMMAs per 12 b128 loads.
// ---------------------------------------------------------------------------

typedef __attribute__((ext_vector_type(16))) _Float16 v16h;
typedef __attribute__((ext_vector_type(8)))  _Float16 v8h;
typedef __attribute__((ext_vector_type(8)))  float    v8f;

static constexpr int N_ROWS = 4096;          // tokens
static constexpr int H_DIM  = 1024;          // model dim
static constexpr int E_DIM  = 2048;          // expansion
static constexpr int S_DIM  = 128;           // shared qk dim
static constexpr int UV_N   = 2 * E_DIM + S_DIM;  // 4224

union Frag16 { v16h v; v8h h[2]; };
struct Acc8 { v8f a[2][4]; };   // [m-subtile][n-subtile]

// Load a 16x32 f16 fragment (A-operand layout per CDNA5 ISA 7.12.2) from a
// row-major matrix with leading dimension `ld`. The same layout serves the
// B-operand when B = W^T with W stored row-major (N x K): lane n walks W's
// row n contiguously along K.
__device__ __forceinline__ v16h load_frag(const _Float16* __restrict__ base,
                                          int row0, int ld, int k0) {
  const int lane = threadIdx.x & 31;
  const _Float16* p = base + (size_t)(row0 + (lane & 15)) * ld
                           + (k0 + ((lane & 16) ? 8 : 0));
  Frag16 f;
  f.h[0] = *(const v8h*)(p);        // K = k0..k0+7   (or +8..+15 for hi lanes)
  f.h[1] = *(const v8h*)(p + 16);   // K = k0+16..+23 (or +24..+31)
  return f.v;
}

__device__ __forceinline__ v8f wmma_f16(v16h a, v16h b, v8f c) {
  // D = A x B + C, f32 accumulate. 8 args: neg_a, A, neg_b, B, c_mod, C, reuse_a, reuse_b
  return __builtin_amdgcn_wmma_f32_16x16x32_f16(false, a, false, b,
                                                (short)0, c, false, false);
}

// One wave computes a 32(M) x 64(N) tile of A(MxK) @ W(NxK)^T.
// Per k-step: 2 A fragments + 4 B fragments -> 8 WMMAs (B reused across both
// M subtiles in registers; A reused across all 4 N subtiles).
__device__ __forceinline__ Acc8 wave_gemm_32x64(const _Float16* __restrict__ A,
                                                const _Float16* __restrict__ W,
                                                int K, int m0, int n0) {
  Acc8 r = {};
  for (int k0 = 0; k0 < K; k0 += 32) {
    if (k0 + 32 < K) {  // gfx1250 global_prefetch_b8 for the next A slabs
      __builtin_prefetch(A + (size_t)(m0 + (threadIdx.x & 15)) * K + k0 + 32, 0, 1);
      __builtin_prefetch(A + (size_t)(m0 + 16 + (threadIdx.x & 15)) * K + k0 + 32, 0, 1);
    }
    v16h a0 = load_frag(A, m0,      K, k0);
    v16h a1 = load_frag(A, m0 + 16, K, k0);
#pragma unroll
    for (int j = 0; j < 4; ++j) {
      v16h b = load_frag(W, n0 + 16 * j, K, k0);
      r.a[0][j] = wmma_f16(a0, b, r.a[0][j]);
      r.a[1][j] = wmma_f16(a1, b, r.a[1][j]);
    }
  }
  return r;
}

// C/D layout (16x16 f32, 8 VGPRs): lane<16 -> rows 0..7, col=lane;
// lane>=16 -> rows 8..15, col=lane-16.
#define EPILOGUE_COORDS()                    \
  const int lane = threadIdx.x & 31;         \
  const int colL = lane & 15;                \
  const int rb   = (lane & 16) ? 8 : 0;

// Tile mapping: m varies fastest across waves so the 8 waves of a block share
// the same n0 (their B fragments hit WGP$ after the first wave).
#define TILE_COORDS(Ncols)                                              \
  const int tiles_m = N_ROWS / 32;                                      \
  const int wave = (blockIdx.x * blockDim.x + threadIdx.x) >> 5;        \
  const int m0 = (wave % tiles_m) * 32;                                 \
  const int n0 = (wave / tiles_m) * 64;                                 \
  if (n0 >= (Ncols)) return; /* wave-uniform guard */

// ---------------------------------------------------------------------------
// Kernel 1: LayerNorm -> f16
// ---------------------------------------------------------------------------
__global__ void gau_layernorm_f16(const float* __restrict__ x,
                                  const float* __restrict__ gamma,
                                  const float* __restrict__ beta,
                                  _Float16* __restrict__ h16) {
  const int row = blockIdx.x;
  const float* xr = x + (size_t)row * H_DIM;
  float s = 0.f, s2 = 0.f;
  for (int i = threadIdx.x; i < H_DIM; i += 256) {
    float v = xr[i];
    s += v; s2 += v * v;
  }
  __shared__ float red0[256], red1[256];
  red0[threadIdx.x] = s; red1[threadIdx.x] = s2;
  __syncthreads();
  for (int off = 128; off > 0; off >>= 1) {
    if (threadIdx.x < off) {
      red0[threadIdx.x] += red0[threadIdx.x + off];
      red1[threadIdx.x] += red1[threadIdx.x + off];
    }
    __syncthreads();
  }
  const float mu  = red0[0] * (1.f / H_DIM);
  const float var = red1[0] * (1.f / H_DIM) - mu * mu;
  const float rs  = rsqrtf(var + 1e-5f);
  for (int i = threadIdx.x; i < H_DIM; i += 256) {
    float v = (xr[i] - mu) * rs * gamma[i] + beta[i];
    h16[(size_t)row * H_DIM + i] = (_Float16)v;
  }
}

// ---------------------------------------------------------------------------
// Kernel 2: f32 -> f16 conversion (weights), 4 elements per thread
// ---------------------------------------------------------------------------
__global__ void gau_cvt_f16(const float* __restrict__ src,
                            _Float16* __restrict__ dst, int n4) {
  int i = blockIdx.x * 256 + threadIdx.x;
  if (i < n4) {
    float4 v = ((const float4*)src)[i];
    v8h* d = (v8h*)dst;           // write 4 halves via half of a v8h slot
    _Float16* dp = dst + (size_t)i * 4;
    dp[0] = (_Float16)v.x; dp[1] = (_Float16)v.y;
    dp[2] = (_Float16)v.z; dp[3] = (_Float16)v.w;
    (void)d;
  }
}

// ---------------------------------------------------------------------------
// Kernel 3: uv = silu(h @ uv_w^T + uv_b); route to u16 / vT16 / q16,k16
// M=4096, N=4224, K=1024
// ---------------------------------------------------------------------------
__global__ void gau_uv_gemm(const _Float16* __restrict__ h16,
                            const _Float16* __restrict__ wuv,
                            const float* __restrict__ uv_b,
                            const float* __restrict__ qk_w,   // (2,128)
                            const float* __restrict__ qk_b,   // (2,128)
                            _Float16* __restrict__ u16,       // (4096,2048)
                            _Float16* __restrict__ vT16,      // (2048,4096)
                            _Float16* __restrict__ q16,       // (4096,128)
                            _Float16* __restrict__ k16) {     // (4096,128)
  TILE_COORDS(UV_N);
  Acc8 acc = wave_gemm_32x64(h16, wuv, H_DIM, m0, n0);
  EPILOGUE_COORDS();
#pragma unroll
  for (int i = 0; i < 2; ++i) {
#pragma unroll
    for (int j = 0; j < 4; ++j) {
      const int ncol = n0 + 16 * j + colL;
      const float bias = uv_b[ncol];
#pragma unroll
      for (int r = 0; r < 8; ++r) {
        const int row = m0 + 16 * i + rb + r;
        float val = acc.a[i][j][r] + bias;
        val = val / (1.f + __expf(-val));  // SiLU
        if (ncol < E_DIM) {
          u16[(size_t)row * E_DIM + ncol] = (_Float16)val;
        } else if (ncol < 2 * E_DIM) {
          // store v transposed so kernel@v sees contiguous K
          vT16[(size_t)(ncol - E_DIM) * N_ROWS + row] = (_Float16)val;
        } else {
          const int s = ncol - 2 * E_DIM;
          q16[(size_t)row * S_DIM + s] = (_Float16)(val * qk_w[s]         + qk_b[s]);
          k16[(size_t)row * S_DIM + s] = (_Float16)(val * qk_w[S_DIM + s] + qk_b[S_DIM + s]);
        }
      }
    }
  }
}

// ---------------------------------------------------------------------------
// Kernel 4: kerm = relu(q @ k^T / sqrt(128))^2   M=4096, N=4096, K=128
// ---------------------------------------------------------------------------
__global__ void gau_qk_gemm(const _Float16* __restrict__ q16,
                            const _Float16* __restrict__ k16,
                            _Float16* __restrict__ kerm) {
  TILE_COORDS(N_ROWS);
  Acc8 acc = wave_gemm_32x64(q16, k16, S_DIM, m0, n0);
  EPILOGUE_COORDS();
#pragma unroll
  for (int i = 0; i < 2; ++i) {
#pragma unroll
    for (int j = 0; j < 4; ++j) {
      const int col = n0 + 16 * j + colL;
#pragma unroll
      for (int r = 0; r < 8; ++r) {
        const int row = m0 + 16 * i + rb + r;
        float v = acc.a[i][j][r];
        v = (v > 0.f) ? v * v * (1.f / 128.f) : 0.f;  // relu(x/sqrt(128))^2
        kerm[(size_t)row * N_ROWS + col] = (_Float16)v;
      }
    }
  }
}

// ---------------------------------------------------------------------------
// Kernel 5: attn = u * (kerm @ v)   M=4096, N=2048, K=4096  (B = vT16 rows)
// ---------------------------------------------------------------------------
__global__ void gau_kv_gemm(const _Float16* __restrict__ kerm,
                            const _Float16* __restrict__ vT16,
                            const _Float16* __restrict__ u16,
                            _Float16* __restrict__ attn16) {
  TILE_COORDS(E_DIM);
  Acc8 acc = wave_gemm_32x64(kerm, vT16, N_ROWS, m0, n0);
  EPILOGUE_COORDS();
#pragma unroll
  for (int i = 0; i < 2; ++i) {
#pragma unroll
    for (int j = 0; j < 4; ++j) {
      const int col = n0 + 16 * j + colL;
#pragma unroll
      for (int r = 0; r < 8; ++r) {
        const int row = m0 + 16 * i + rb + r;
        float g = (float)u16[(size_t)row * E_DIM + col];
        attn16[(size_t)row * E_DIM + col] = (_Float16)(acc.a[i][j][r] * g);
      }
    }
  }
}

// ---------------------------------------------------------------------------
// Kernel 6: out = attn @ o_w^T + o_b + shortcut   M=4096, N=1024, K=2048
// ---------------------------------------------------------------------------
__global__ void gau_out_gemm(const _Float16* __restrict__ attn16,
                             const _Float16* __restrict__ wo,
                             const float* __restrict__ o_b,
                             const float* __restrict__ x,
                             float* __restrict__ out) {
  TILE_COORDS(H_DIM);
  Acc8 acc = wave_gemm_32x64(attn16, wo, E_DIM, m0, n0);
  EPILOGUE_COORDS();
#pragma unroll
  for (int i = 0; i < 2; ++i) {
#pragma unroll
    for (int j = 0; j < 4; ++j) {
      const int col = n0 + 16 * j + colL;
      const float bias = o_b[col];
#pragma unroll
      for (int r = 0; r < 8; ++r) {
        const int row = m0 + 16 * i + rb + r;
        out[(size_t)row * H_DIM + col] =
            acc.a[i][j][r] + bias + x[(size_t)row * H_DIM + col];
      }
    }
  }
}

// ---------------------------------------------------------------------------
// Host launcher
// ---------------------------------------------------------------------------
extern "C" void kernel_launch(void* const* d_in, const int* in_sizes, int n_in,
                              void* d_out, int out_size, void* d_ws, size_t ws_size,
                              hipStream_t stream) {
  const float* x     = (const float*)d_in[0];
  const float* ln_g  = (const float*)d_in[1];
  const float* ln_b  = (const float*)d_in[2];
  const float* uv_w  = (const float*)d_in[3];
  const float* uv_b  = (const float*)d_in[4];
  const float* qk_w  = (const float*)d_in[5];
  const float* qk_b  = (const float*)d_in[6];
  const float* o_w   = (const float*)d_in[7];
  const float* o_b   = (const float*)d_in[8];
  float* out = (float*)d_out;

  // carve workspace (all f16 intermediates)
  char* ws = (char*)d_ws;
  auto carve = [&](size_t elems) {
    _Float16* p = (_Float16*)ws;
    ws += (elems * sizeof(_Float16) + 255) & ~(size_t)255;
    return p;
  };
  _Float16* h16    = carve((size_t)N_ROWS * H_DIM);      // 8.4 MB
  _Float16* wuv16  = carve((size_t)UV_N * H_DIM);        // 8.7 MB
  _Float16* wo16   = carve((size_t)H_DIM * E_DIM);       // 4.2 MB
  _Float16* u16    = carve((size_t)N_ROWS * E_DIM);      // 16.8 MB
  _Float16* vT16   = carve((size_t)E_DIM * N_ROWS);      // 16.8 MB
  _Float16* q16    = carve((size_t)N_ROWS * S_DIM);      // 1 MB
  _Float16* k16    = carve((size_t)N_ROWS * S_DIM);      // 1 MB
  _Float16* kerm16 = carve((size_t)N_ROWS * N_ROWS);     // 33.6 MB
  _Float16* attn16 = carve((size_t)N_ROWS * E_DIM);      // 16.8 MB

  // 1. LayerNorm -> f16
  gau_layernorm_f16<<<N_ROWS, 256, 0, stream>>>(x, ln_g, ln_b, h16);

  // 2. weight conversions (vectorized x4)
  {
    int n1 = UV_N * H_DIM / 4;
    gau_cvt_f16<<<(n1 + 255) / 256, 256, 0, stream>>>(uv_w, wuv16, n1);
    int n2 = H_DIM * E_DIM / 4;
    gau_cvt_f16<<<(n2 + 255) / 256, 256, 0, stream>>>(o_w, wo16, n2);
  }

  // 3. uv GEMM + SiLU + split   (128 * 66 wave-tiles -> 1056 blocks)
  gau_uv_gemm<<<(N_ROWS / 32) * (UV_N / 64) / 8, 256, 0, stream>>>(
      h16, wuv16, uv_b, qk_w, qk_b, u16, vT16, q16, k16);

  // 4. qk GEMM + relu^2         (128 * 64 wave-tiles -> 1024 blocks)
  gau_qk_gemm<<<(N_ROWS / 32) * (N_ROWS / 64) / 8, 256, 0, stream>>>(
      q16, k16, kerm16);

  // 5. kernel @ v, gated by u   (128 * 32 wave-tiles -> 512 blocks)
  gau_kv_gemm<<<(N_ROWS / 32) * (E_DIM / 64) / 8, 256, 0, stream>>>(
      kerm16, vT16, u16, attn16);

  // 6. output projection + bias + residual (128 * 16 wave-tiles -> 256 blocks)
  gau_out_gemm<<<(N_ROWS / 32) * (H_DIM / 64) / 8, 256, 0, stream>>>(
      attn16, wo16, o_b, x, out);
}
